// VQCodebook_86921548137127
// MI455X (gfx1250) — matrix operand order repaired and checked by reading, not compiled, not run on hardware
//
#include <hip/hip_runtime.h>

typedef __attribute__((ext_vector_type(16))) _Float16     v16h;
typedef __attribute__((ext_vector_type(8)))  float        v8f;
typedef __attribute__((ext_vector_type(8)))  unsigned int v8u;

#define DDIM 64
#define KCB  512
#define ROWS_TOTAL  (8 * 16384)
#define ROWS_PER_WG 256                      // 8 waves x 2 blocks x 16 rows
#define LDS_BYTES   (65536 + 65536 + 2048)   // cb_hi + cb_lo + e_sq

static __device__ __forceinline__ unsigned int pack2(float a, float b) {
  unsigned short ua = __builtin_bit_cast(unsigned short, (_Float16)a);
  unsigned short ub = __builtin_bit_cast(unsigned short, (_Float16)b);
  return (unsigned int)ua | ((unsigned int)ub << 16);
}

__global__ __launch_bounds__(256) void vq_f16x3_kernel(
    const float* __restrict__ x, const float* __restrict__ cb,
    float* __restrict__ out) {
  extern __shared__ unsigned char smem[];
  unsigned int* cbh = (unsigned int*)smem;             // [512][32] f16 pairs (hi)
  unsigned int* cbl = (unsigned int*)(smem + 65536);   // [512][32] f16 pairs (lo)
  float*        esq = (float*)(smem + 131072);         // [512]

  // ---- Stage full codebook as f16 hi/lo + e_sq into LDS (once per WG) ----
  for (int e = threadIdx.x; e < KCB; e += 256) {
    const float4* src = (const float4*)(cb + (size_t)e * DDIM);
    float ss = 0.f;
#pragma unroll
    for (int i = 0; i < DDIM / 4; ++i) {
      float4 v = src[i];
      ss += v.x * v.x + v.y * v.y + v.z * v.z + v.w * v.w;
      float hx = (float)(_Float16)v.x, hy = (float)(_Float16)v.y;
      float hz = (float)(_Float16)v.z, hw = (float)(_Float16)v.w;
      cbh[e * 32 + 2 * i    ] = pack2(v.x, v.y);
      cbh[e * 32 + 2 * i + 1] = pack2(v.z, v.w);
      cbl[e * 32 + 2 * i    ] = pack2(v.x - hx, v.y - hy);
      cbl[e * 32 + 2 * i + 1] = pack2(v.z - hz, v.w - hw);
    }
    esq[e] = ss;
  }
  __syncthreads();

  const int lane = threadIdx.x & 31;
  const int wave = threadIdx.x >> 5;
  const int h    = lane >> 4;     // lane-half selects K sub-runs / row half
  const int mr   = lane & 15;
  const size_t rowbase = (size_t)blockIdx.x * ROWS_PER_WG + (size_t)wave * 32;

  // ---- A fragments (f16 hi/lo), 2 row-blocks x 2 K-chunks ----
  // A k order (contiguous, same as B): lane-half h, element e: K = 32*c + 16*h + e
  v16h ah[2][2], al[2][2];
#pragma unroll
  for (int b = 0; b < 2; ++b) {
    const float* xr = x + (rowbase + b * 16 + mr) * DDIM;
#pragma unroll
    for (int c = 0; c < 2; ++c) {
#pragma unroll
      for (int r = 0; r < 2; ++r) {
        const float* p = xr + 32 * c + 16 * h + 8 * r;
        float4 v0 = *(const float4*)(p);
        float4 v1 = *(const float4*)(p + 4);
        float f[8] = {v0.x, v0.y, v0.z, v0.w, v1.x, v1.y, v1.z, v1.w};
#pragma unroll
        for (int j = 0; j < 8; ++j) {
          _Float16 hi = (_Float16)f[j];
          _Float16 lo = (_Float16)(f[j] - (float)hi);
          ah[b][c][r * 8 + j] = hi;
          al[b][c][r * 8 + j] = lo;
        }
      }
    }
  }

  float best[2][8];
  int   bidx[2][8];
#pragma unroll
  for (int b = 0; b < 2; ++b)
#pragma unroll
    for (int i = 0; i < 8; ++i) { best[b][i] = 3.4e38f; bidx[b][i] = 0; }

  // ---- Main loop: 32 codebook tiles of 16 entries ----
  for (int n0 = 0; n0 < KCB; n0 += 16) {
    const int e = n0 + mr;  // this lane's column index
    // B fragments: contiguous 16 f16 at cb[e][32c + 16h ..], same k order as A
    v16h bh[2], bl[2];
#pragma unroll
    for (int c = 0; c < 2; ++c) {
      bh[c] = __builtin_bit_cast(v16h, *(const v8u*)&cbh[e * 32 + 16 * c + 8 * h]);
      bl[c] = __builtin_bit_cast(v16h, *(const v8u*)&cbl[e * 32 + 16 * c + 8 * h]);
    }
    const float es = esq[e];
#pragma unroll
    for (int b = 0; b < 2; ++b) {
      v8f acc = {};
#pragma unroll
      for (int c = 0; c < 2; ++c) {
        acc = __builtin_amdgcn_wmma_f32_16x16x32_f16(
            false, ah[b][c], false, bh[c], (short)0, acc, false, false);
        acc = __builtin_amdgcn_wmma_f32_16x16x32_f16(
            false, ah[b][c], false, bl[c], (short)0, acc, false, false);
        acc = __builtin_amdgcn_wmma_f32_16x16x32_f16(
            false, al[b][c], false, bh[c], (short)0, acc, false, false);
      }
#pragma unroll
      for (int i = 0; i < 8; ++i) {
        float s = fmaf(acc[i], -2.0f, es);     // e^2[k] - 2*cross
        if (s < best[b][i]) { best[b][i] = s; bidx[b][i] = e; }
      }
    }
  }

  // ---- Reduce across 16 lanes per row (xor masks stay within half) ----
#pragma unroll
  for (int b = 0; b < 2; ++b) {
#pragma unroll
    for (int i = 0; i < 8; ++i) {
      float s = best[b][i];
      int   k = bidx[b][i];
#pragma unroll
      for (int m = 1; m <= 8; m <<= 1) {
        float s2 = __shfl_xor(s, m, 32);
        int   k2 = __shfl_xor(k, m, 32);
        if (s2 < s || (s2 == s && k2 < k)) { s = s2; k = k2; }
      }
      bidx[b][i] = k;
    }
  }

  // ---- Gather winning codebook rows (f32 source) to output ----
#pragma unroll
  for (int b = 0; b < 2; ++b) {
#pragma unroll
    for (int i = 0; i < 8; ++i) {
      const size_t row = rowbase + b * 16 + (size_t)h * 8 + i;
      const float4 v =
          *(const float4*)(cb + (size_t)bidx[b][i] * DDIM + mr * 4);
      *(float4*)(out + row * DDIM + mr * 4) = v;
    }
  }
}

// ---------------------------------------------------------------------------
extern "C" void kernel_launch(void* const* d_in, const int* in_sizes, int n_in,
                              void* d_out, int out_size, void* d_ws,
                              size_t ws_size, hipStream_t stream) {
  const float* x  = (const float*)d_in[0];  // [8,16384,64] f32
  const float* cb = (const float*)d_in[1];  // [512,64] f32
  float* out = (float*)d_out;               // [8,16384,1,64] f32

  (void)hipFuncSetAttribute((const void*)vq_f16x3_kernel,
                            hipFuncAttributeMaxDynamicSharedMemorySize,
                            LDS_BYTES);
  vq_f16x3_kernel<<<ROWS_TOTAL / ROWS_PER_WG, 256, LDS_BYTES, stream>>>(x, cb,
                                                                        out);
}
